// EncoderBlock_22789096473204
// MI455X (gfx1250) — matrix operand order repaired and checked by reading, not compiled, not run on hardware
//
#include <hip/hip_runtime.h>
#include <hip/hip_bf16.h>
#include <cstdint>
#include <cstddef>

typedef __attribute__((ext_vector_type(16))) _Float16 v16h;
typedef __attribute__((ext_vector_type(8)))  _Float16 v8h;
typedef __attribute__((ext_vector_type(8)))  float    v8f;

#define B_   512
#define S_   81
#define D_   1024
#define NH_  16
#define DH_  64
#define C_   32
#define H1_  256
#define G_   256
#define FF_  4096
#define MROWS (B_*S_)   /* 41472 */

enum { EPI_BIAS = 0, EPI_SILU = 1, EPI_MISH = 2, EPI_RES = 3, EPI_SCALE = 4 };

__device__ __forceinline__ v16h cat16(v8h lo, v8h hi) {
  return __builtin_shufflevector(lo, hi, 0,1,2,3,4,5,6,7,8,9,10,11,12,13,14,15);
}
__device__ __forceinline__ v8f wmma_f16(v16h a, v16h b, v8f c) {
  return __builtin_amdgcn_wmma_f32_16x16x32_f16(false, a, false, b, (short)0, c, false, false);
}

// CDNA5 async global->LDS copy (ASYNCcnt). LDS operand = low 32 bits of the
// generic shared-memory address.
__device__ __forceinline__ void async_ld_b128(const _Float16* g, _Float16* l) {
  asm volatile("global_load_async_to_lds_b128 %0, %1, off"
               :: "v"((unsigned)(uintptr_t)l),
                  "v"((unsigned long long)(uintptr_t)g)
               : "memory");
}
__device__ __forceinline__ void wait_async0() {
  asm volatile("s_wait_asynccnt 0" ::: "memory");
}

// ---------------- elementwise f32 -> f16 ----------------
__global__ __launch_bounds__(256) void k_cvt_f16(const float* __restrict__ x,
                                                 _Float16* __restrict__ y, size_t n) {
  size_t i = (size_t)blockIdx.x * 256 + threadIdx.x;
  if (i < n) y[i] = (_Float16)x[i];
}

// ---------------- W (K x N) f32  ->  Wt (N x K) f16 ----------------
__global__ __launch_bounds__(256) void k_transpose_f16(const float* __restrict__ w,
                                                       _Float16* __restrict__ wt,
                                                       int K, int N) {
  size_t i = (size_t)blockIdx.x * 256 + threadIdx.x;
  size_t tot = (size_t)K * N;
  if (i < tot) {
    int k = (int)(i / N);
    int n = (int)(i % N);
    wt[(size_t)n * K + k] = (_Float16)w[i];
  }
}

// ---------------- generic WMMA GEMM with async-to-LDS staging ----------------
// C[M,N] = epilogue(A[M,K] @ Bt[N,K]^T + bias [, res])
// A, Bt f16 row-major. M % BM == 0, K % 32 == 0; N guarded.
template<int BM, int BN, int WM, int WN, int EPI>
__global__ __launch_bounds__(256) void k_gemm(
    const _Float16* __restrict__ A, const _Float16* __restrict__ Bt,
    const float* __restrict__ bias, const float* __restrict__ res,
    float* __restrict__ Cf, _Float16* __restrict__ Ch,
    int M, int N, int K, float scale) {
  constexpr int LDP = 40;                 // f16 pitch (80B) -> conflict-friendly
  constexpr int FM  = BM / (16 * WM);
  constexpr int FN  = BN / (16 * WN);
  constexpr int NCA = (BM * 4 + 255) / 256;
  constexpr int NCB = (BN * 4 + 255) / 256;

  __shared__ _Float16 As[2][BM * LDP] __attribute__((aligned(16)));
  __shared__ _Float16 Bs[2][BN * LDP] __attribute__((aligned(16)));

  const int tid  = threadIdx.x;
  const int wave = tid >> 5, lane = tid & 31;
  const int lr = lane & 15, lh = lane >> 4;
  const int wm = wave % WM, wn = wave / WM;
  const int m0 = blockIdx.y * BM;
  const int n0 = blockIdx.x * BN;
  const int wmo = wm * FM * 16;
  const int wno = wn * FN * 16;
  const v8h vzero = {};

  // issue async copies of one K-chunk into buffer `buf`
  auto issue_stage = [&](int kc, int buf) {
#pragma unroll
    for (int i = 0; i < NCA; ++i) {
      int c = tid + i * 256;
      if (c < BM * 4) {
        int r = c >> 2, q = (c & 3) * 8;
        async_ld_b128(A + (size_t)(m0 + r) * K + kc * 32 + q, &As[buf][r * LDP + q]);
      }
    }
#pragma unroll
    for (int i = 0; i < NCB; ++i) {
      int c = tid + i * 256;
      if (c < BN * 4) {
        int r = c >> 2, q = (c & 3) * 8;
        int gn = n0 + r;
        if (gn < N)
          async_ld_b128(Bt + (size_t)gn * K + kc * 32 + q, &Bs[buf][r * LDP + q]);
      }
    }
  };

  // zero-fill out-of-range B rows once (never async-written -> stay zero)
#pragma unroll
  for (int i = 0; i < NCB; ++i) {
    int c = tid + i * 256;
    if (c < BN * 4) {
      int r = c >> 2, q = (c & 3) * 8;
      if (n0 + r >= N) {
        *(v8h*)&Bs[0][r * LDP + q] = vzero;
        *(v8h*)&Bs[1][r * LDP + q] = vzero;
      }
    }
  }

  v8f acc[FM][FN] = {};

  const int nk = K >> 5;
  issue_stage(0, 0);
  wait_async0();
  __syncthreads();

  for (int c = 0; c < nk; ++c) {
    if (c + 1 < nk) issue_stage(c + 1, (c + 1) & 1);   // hide latency under WMMAs
    const int cur = c & 1;
    v16h af[FM], bf[FN];
#pragma unroll
    for (int fm = 0; fm < FM; ++fm) {
      const _Float16* p = &As[cur][(wmo + fm * 16 + lr) * LDP + lh * 8];
      af[fm] = cat16(*(const v8h*)p, *(const v8h*)(p + 16));
    }
#pragma unroll
    for (int fn = 0; fn < FN; ++fn) {
      const _Float16* p = &Bs[cur][(wno + fn * 16 + lr) * LDP + lh * 8];
      bf[fn] = cat16(*(const v8h*)p, *(const v8h*)(p + 16));
    }
#pragma unroll
    for (int fm = 0; fm < FM; ++fm)
#pragma unroll
      for (int fn = 0; fn < FN; ++fn)
        acc[fm][fn] = wmma_f16(af[fm], bf[fn], acc[fm][fn]);
    wait_async0();
    __syncthreads();
  }

  // epilogue: C VGPR r -> row = r + 8*(lane>>4), col = lane&15
#pragma unroll
  for (int fm = 0; fm < FM; ++fm) {
#pragma unroll
    for (int fn = 0; fn < FN; ++fn) {
#pragma unroll
      for (int r = 0; r < 8; ++r) {
        int gm = m0 + wmo + fm * 16 + lh * 8 + r;
        int gn = n0 + wno + fn * 16 + lr;
        if (gn < N) {
          float v = acc[fm][fn][r] + bias[gn];
          if (EPI == EPI_SILU) {
            v = v * (1.0f / (1.0f + __expf(-v)));
          } else if (EPI == EPI_MISH) {
            float sp = (v > 20.0f) ? v : log1pf(__expf(v));
            v = v * tanhf(sp);
          } else if (EPI == EPI_RES) {
            v += res[(size_t)gm * N + gn];
          } else if (EPI == EPI_SCALE) {
            v *= scale;
          }
          if (Cf) Cf[(size_t)gm * N + gn] = v;
          if (Ch) Ch[(size_t)gm * N + gn] = (_Float16)v;
        }
      }
    }
  }
}

// ---------------- row LayerNorm (+ optional SiLU), f32 in, f32/f16 out ----------------
template<int ACT>
__global__ __launch_bounds__(256) void k_row_ln(const float* __restrict__ X,
                                                const float* __restrict__ g,
                                                const float* __restrict__ b,
                                                float* __restrict__ outF,
                                                _Float16* __restrict__ outH, int N) {
  const int row = blockIdx.x;
  const float* x = X + (size_t)row * N;
  const int tid = threadIdx.x;
  float s = 0.0f, s2 = 0.0f;
  for (int j = tid; j < N; j += 256) { float v = x[j]; s += v; s2 += v * v; }
  __shared__ float r1[256], r2[256];
  r1[tid] = s; r2[tid] = s2;
  __syncthreads();
  for (int off = 128; off > 0; off >>= 1) {
    if (tid < off) { r1[tid] += r1[tid + off]; r2[tid] += r2[tid + off]; }
    __syncthreads();
  }
  float mean = r1[0] / N;
  float var  = r2[0] / N - mean * mean;
  float inv  = rsqrtf(var + 1e-5f);
  for (int j = tid; j < N; j += 256) {
    float y = (x[j] - mean) * inv * g[j] + b[j];
    if (ACT == 1) y = y * (1.0f / (1.0f + __expf(-y)));
    if (outF) outF[(size_t)row * N + j] = y;
    if (outH) outH[(size_t)row * N + j] = (_Float16)y;
  }
}

// ---------------- fused attention per (b, h): softmax(q k^T * s + bias) v ----------------
// q already scaled by 1/sqrt(DH). S=81 padded to 96. Dynamic LDS: 73728 B.
__global__ __launch_bounds__(256) void k_attn(const _Float16* __restrict__ qg,
                                              const _Float16* __restrict__ kg,
                                              const _Float16* __restrict__ vg,
                                              const float* __restrict__ smol,
                                              _Float16* __restrict__ out) {
  const int bh = blockIdx.x;
  const int b  = bh >> 4;
  const int h  = bh & 15;
  extern __shared__ char smem[];
  _Float16* qs  = (_Float16*)smem;          // 96 x 64
  _Float16* ks  = qs + 96 * 64;             // 96 x 64
  _Float16* vts = ks + 96 * 64;             // 64 x 96 (transposed)
  float*    ss  = (float*)(vts + 64 * 96);  // 96 x 96: bias -> scores -> f16 probs

  const int tid = threadIdx.x;
  const int wave = tid >> 5, lane = tid & 31;
  const int lr = lane & 15, lh = lane >> 4;
  const v8h vzero = {};

  // zero padded rows of q/k tiles (rows 81..95), then async-stage real rows
  for (int idx = tid; idx < (96 - S_) * 8; idx += 256) {
    int i = S_ + (idx >> 3), dc = (idx & 7) * 8;
    *(v8h*)&qs[i * 64 + dc] = vzero;
    *(v8h*)&ks[i * 64 + dc] = vzero;
  }
  for (int idx = tid; idx < S_ * 8; idx += 256) {
    int i = idx >> 3, dc = (idx & 7) * 8;
    size_t gaddr = ((size_t)(b * S_ + i)) * D_ + h * DH_ + dc;
    async_ld_b128(qg + gaddr, &qs[i * 64 + dc]);
    async_ld_b128(kg + gaddr, &ks[i * 64 + dc]);
  }
  // v, transposed into LDS (unconditional clamped loads, select for pads)
  for (int idx = tid; idx < 96 * 8; idx += 256) {
    int i = idx >> 3, dc = (idx & 7) * 8;
    int ic = (i < S_) ? i : (S_ - 1);
    size_t gaddr = ((size_t)(b * S_ + ic)) * D_ + h * DH_ + dc;
    v8h vv = *(const v8h*)(vg + gaddr);
    if (i >= S_) vv = vzero;
#pragma unroll
    for (int j = 0; j < 8; ++j) vts[(dc + j) * 96 + i] = vv[j];
  }
  // smolgen bias tile into ss (clamped unconditional loads; -inf pads)
  const float* srow = smol + (size_t)bh * (S_ * S_);
  for (int idx = tid; idx < 96 * 96; idx += 256) {
    int i = idx / 96, j = idx - i * 96;
    int ic = (i < S_) ? i : (S_ - 1);
    int jc = (j < S_) ? j : (S_ - 1);
    float v = srow[ic * S_ + jc];
    ss[idx] = (i < S_ && j < S_) ? v : -1e30f;
  }
  wait_async0();
  __syncthreads();

  // scores: 6x6 tiles, K = 64 (2 WMMA steps); accumulate onto staged bias in LDS
  for (int t = wave; t < 36; t += 8) {
    int ti = t / 6, tj = t % 6;
    v8f acc = {};
#pragma unroll
    for (int kk = 0; kk < 2; ++kk) {
      const _Float16* pa = &qs[(ti * 16 + lr) * 64 + kk * 32 + lh * 8];
      v16h a = cat16(*(const v8h*)pa, *(const v8h*)(pa + 16));
      const _Float16* pb = &ks[(tj * 16 + lr) * 64 + kk * 32 + lh * 8];
      v16h bm = cat16(*(const v8h*)pb, *(const v8h*)(pb + 16));
      acc = wmma_f16(a, bm, acc);
    }
#pragma unroll
    for (int r = 0; r < 8; ++r) {
      int i = ti * 16 + lh * 8 + r;
      int j = tj * 16 + lr;
      ss[i * 96 + j] += acc[r];
    }
  }
  __syncthreads();

  // softmax, one row per thread; convert to f16 in place
  if (tid < 96) {
    float* r = ss + tid * 96;
    _Float16* p = (_Float16*)r;
    if (tid < S_) {
      float mx = -1e30f;
      for (int j = 0; j < S_; ++j) mx = fmaxf(mx, r[j]);
      float sum = 0.0f;
      for (int j = 0; j < S_; ++j) sum += __expf(r[j] - mx);
      float rinv = 1.0f / sum;
      for (int j = 0; j < 96; ++j) {
        float e = (j < S_) ? __expf(r[j] - mx) * rinv : 0.0f;
        p[j] = (_Float16)e;   // f16 slot j lands in float word j/2 (already consumed)
      }
    } else {
      for (int j = 0; j < 96; ++j) p[j] = (_Float16)0.0f;
    }
  }
  __syncthreads();

  // out = P @ V : 6x4 tiles, K = 96 (3 WMMA steps)
  for (int t = wave; t < 24; t += 8) {
    int ti = t >> 2, tj = t & 3;
    v8f acc = {};
#pragma unroll
    for (int kk = 0; kk < 3; ++kk) {
      const _Float16* pa = (const _Float16*)(ss + (ti * 16 + lr) * 96) + kk * 32 + lh * 8;
      v16h a = cat16(*(const v8h*)pa, *(const v8h*)(pa + 16));
      const _Float16* pb = &vts[(tj * 16 + lr) * 96 + kk * 32 + lh * 8];
      v16h bm = cat16(*(const v8h*)pb, *(const v8h*)(pb + 16));
      acc = wmma_f16(a, bm, acc);
    }
#pragma unroll
    for (int r = 0; r < 8; ++r) {
      int i = ti * 16 + lh * 8 + r;
      int d = tj * 16 + lr;
      if (i < S_) out[((size_t)(b * S_ + i)) * D_ + h * DH_ + d] = (_Float16)acc[r];
    }
  }
}

// ---------------- host orchestration ----------------
extern "C" void kernel_launch(void* const* d_in, const int* in_sizes, int n_in,
                              void* d_out, int out_size, void* d_ws, size_t ws_size,
                              hipStream_t stream) {
  (void)in_sizes; (void)n_in; (void)out_size; (void)ws_size;
  const float* x      = (const float*)d_in[0];
  const float* w_c    = (const float*)d_in[1];
  const float* b_c    = (const float*)d_in[2];
  const float* w_d1   = (const float*)d_in[3];
  const float* b_d1   = (const float*)d_in[4];
  const float* ln1s_g = (const float*)d_in[5];
  const float* ln1s_b = (const float*)d_in[6];
  const float* w_d2   = (const float*)d_in[7];
  const float* b_d2   = (const float*)d_in[8];
  const float* ln2s_g = (const float*)d_in[9];
  const float* ln2s_b = (const float*)d_in[10];
  const float* w_gg   = (const float*)d_in[11];
  const float* b_gg   = (const float*)d_in[12];
  const float* wq     = (const float*)d_in[13];
  const float* bq     = (const float*)d_in[14];
  const float* wk     = (const float*)d_in[15];
  const float* bk     = (const float*)d_in[16];
  const float* wv     = (const float*)d_in[17];
  const float* bv     = (const float*)d_in[18];
  const float* wo     = (const float*)d_in[19];
  const float* bo     = (const float*)d_in[20];
  const float* ln1_g  = (const float*)d_in[21];
  const float* ln1_b  = (const float*)d_in[22];
  const float* w_f1   = (const float*)d_in[23];
  const float* b_f1   = (const float*)d_in[24];
  const float* w_f2   = (const float*)d_in[25];
  const float* b_f2   = (const float*)d_in[26];
  const float* fln_g  = (const float*)d_in[27];
  const float* fln_b  = (const float*)d_in[28];
  float* out = (float*)d_out;

  char* ws = (char*)d_ws;
  size_t off = 0;
  auto alloc = [&](size_t bytes) -> char* {
    char* p = ws + off;
    off += (bytes + 255) & ~(size_t)255;
    return p;
  };

  _Float16* xh   = (_Float16*)alloc((size_t)MROWS * D_ * 2);
  _Float16* wct  = (_Float16*)alloc((size_t)C_ * D_ * 2);
  _Float16* wd1t = (_Float16*)alloc((size_t)H1_ * (S_ * C_) * 2);
  _Float16* wd2t = (_Float16*)alloc((size_t)(NH_ * G_) * H1_ * 2);
  _Float16* wggt = (_Float16*)alloc((size_t)(S_ * S_) * G_ * 2);
  _Float16* wqt  = (_Float16*)alloc((size_t)D_ * D_ * 2);
  _Float16* wkt  = (_Float16*)alloc((size_t)D_ * D_ * 2);
  _Float16* wvt  = (_Float16*)alloc((size_t)D_ * D_ * 2);
  _Float16* wot  = (_Float16*)alloc((size_t)D_ * D_ * 2);
  _Float16* wf1t = (_Float16*)alloc((size_t)FF_ * D_ * 2);
  _Float16* wf2t = (_Float16*)alloc((size_t)D_ * FF_ * 2);
  _Float16* h0h  = (_Float16*)alloc((size_t)MROWS * C_ * 2);
  float*    t1   = (float*)alloc((size_t)B_ * H1_ * 4);
  _Float16* h1h  = (_Float16*)alloc((size_t)B_ * H1_ * 2);
  float*    t2   = (float*)alloc((size_t)B_ * (NH_ * G_) * 4);
  _Float16* h2h  = (_Float16*)alloc((size_t)B_ * (NH_ * G_) * 2);
  float*    smol = (float*)alloc((size_t)B_ * NH_ * S_ * S_ * 4);
  _Float16* qhB  = (_Float16*)alloc((size_t)MROWS * D_ * 2);
  _Float16* khB  = (_Float16*)alloc((size_t)MROWS * D_ * 2);
  _Float16* vhB  = (_Float16*)alloc((size_t)MROWS * D_ * 2);
  _Float16* ao   = (_Float16*)alloc((size_t)MROWS * D_ * 2);
  float*    hres = (float*)alloc((size_t)MROWS * D_ * 4);
  float*    hf   = (float*)alloc((size_t)MROWS * D_ * 4);
  _Float16* hh   = (_Float16*)alloc((size_t)MROWS * D_ * 2);
  _Float16* fh   = (_Float16*)alloc((size_t)MROWS * FF_ * 2);
  float*    t3   = (float*)alloc((size_t)MROWS * D_ * 4);

  auto cvt = [&](const float* src, _Float16* dst, size_t n) {
    k_cvt_f16<<<(unsigned)((n + 255) / 256), 256, 0, stream>>>(src, dst, n);
  };
  auto tr = [&](const float* w, _Float16* wt, int K, int N) {
    size_t n = (size_t)K * N;
    k_transpose_f16<<<(unsigned)((n + 255) / 256), 256, 0, stream>>>(w, wt, K, N);
  };

  // --- precision conversion / weight transposes ---
  cvt(x, xh, (size_t)MROWS * D_);
  tr(w_c,  wct,  D_,      C_);
  tr(w_d1, wd1t, S_ * C_, H1_);
  tr(w_d2, wd2t, H1_,     NH_ * G_);
  tr(w_gg, wggt, G_,      S_ * S_);
  tr(wq,   wqt,  D_, D_);
  tr(wk,   wkt,  D_, D_);
  tr(wv,   wvt,  D_, D_);
  tr(wo,   wot,  D_, D_);
  tr(w_f1, wf1t, D_, FF_);
  tr(w_f2, wf2t, FF_, D_);

  // --- smolgen ---
  k_gemm<128, 32, 8, 1, EPI_SILU><<<dim3(1, MROWS / 128), 256, 0, stream>>>(
      xh, wct, b_c, nullptr, nullptr, h0h, MROWS, C_, D_, 1.0f);
  k_gemm<128, 128, 4, 2, EPI_BIAS><<<dim3(H1_ / 128, B_ / 128), 256, 0, stream>>>(
      h0h, wd1t, b_d1, nullptr, t1, nullptr, B_, H1_, S_ * C_, 1.0f);
  k_row_ln<1><<<B_, 256, 0, stream>>>(t1, ln1s_g, ln1s_b, nullptr, h1h, H1_);
  k_gemm<128, 128, 4, 2, EPI_BIAS><<<dim3((NH_ * G_) / 128, B_ / 128), 256, 0, stream>>>(
      h1h, wd2t, b_d2, nullptr, t2, nullptr, B_, NH_ * G_, H1_, 1.0f);
  k_row_ln<1><<<B_, 256, 0, stream>>>(t2, ln2s_g, ln2s_b, nullptr, h2h, NH_ * G_);
  k_gemm<128, 128, 4, 2, EPI_BIAS><<<dim3((S_ * S_ + 127) / 128, (B_ * NH_) / 128), 256, 0, stream>>>(
      h2h, wggt, b_gg, nullptr, smol, nullptr, B_ * NH_, S_ * S_, G_, 1.0f);

  // --- q, k, v projections (q pre-scaled by 1/sqrt(DH)) ---
  k_gemm<128, 128, 4, 2, EPI_SCALE><<<dim3(D_ / 128, MROWS / 128), 256, 0, stream>>>(
      xh, wqt, bq, nullptr, nullptr, qhB, MROWS, D_, D_, 0.125f);
  k_gemm<128, 128, 4, 2, EPI_BIAS><<<dim3(D_ / 128, MROWS / 128), 256, 0, stream>>>(
      xh, wkt, bk, nullptr, nullptr, khB, MROWS, D_, D_, 1.0f);
  k_gemm<128, 128, 4, 2, EPI_BIAS><<<dim3(D_ / 128, MROWS / 128), 256, 0, stream>>>(
      xh, wvt, bv, nullptr, nullptr, vhB, MROWS, D_, D_, 1.0f);

  // --- fused attention with smolgen bias ---
  k_attn<<<B_ * NH_, 256, 73728, stream>>>(qhB, khB, vhB, smol, ao);

  // --- output projection + residual + LN ---
  k_gemm<128, 128, 4, 2, EPI_RES><<<dim3(D_ / 128, MROWS / 128), 256, 0, stream>>>(
      ao, wot, bo, x, hres, nullptr, MROWS, D_, D_, 1.0f);
  k_row_ln<0><<<MROWS, 256, 0, stream>>>(hres, ln1_g, ln1_b, hf, hh, D_);

  // --- FFN + residual + LN ---
  k_gemm<128, 128, 4, 2, EPI_MISH><<<dim3(FF_ / 128, MROWS / 128), 256, 0, stream>>>(
      hh, wf1t, b_f1, nullptr, nullptr, fh, MROWS, FF_, D_, 1.0f);
  k_gemm<128, 128, 4, 2, EPI_RES><<<dim3(D_ / 128, MROWS / 128), 256, 0, stream>>>(
      fh, wf2t, b_f2, hf, t3, nullptr, MROWS, D_, FF_, 1.0f);
  k_row_ln<0><<<MROWS, 256, 0, stream>>>(t3, fln_g, fln_b, out, nullptr, D_);
}
